// DCN_89060441850476
// MI455X (gfx1250) — compile-verified, matrix operand order checked by reading, not run on hardware
//
#include <hip/hip_runtime.h>
#include <math.h>

// Problem constants (from reference)
#define NB   8
#define HH   64
#define WW   64
#define CC   256
#define GG   32
#define GCH  8
#define OMN  864           // G * KK * 3
#define NHW  (NB*HH*WW)    // 32768

typedef __attribute__((ext_vector_type(16))) __bf16 v16bf;
typedef __attribute__((ext_vector_type(8)))  __bf16 v8bf;
typedef __attribute__((ext_vector_type(8)))  float  v8f;

__device__ __forceinline__ void split_bf16(float x, __bf16& hi, __bf16& lo) {
    hi = (__bf16)x;
    lo = (__bf16)(x - (float)hi);
}

__device__ __forceinline__ v16bf cat16(v8bf a, v8bf b) {
    v16bf r;
#pragma unroll
    for (int i = 0; i < 8; ++i) { r[i] = a[i]; r[i + 8] = b[i]; }
    return r;
}

__device__ __forceinline__ v16bf load_frag(const __bf16* p) {
    // ISA 16-bit A/B fragment: v0..3 = K kb..kb+7, v4..7 = K kb+16..kb+23
    return cat16(*(const v8bf*)p, *(const v8bf*)(p + 16));
}

// Split-bf16 fp32 emulation: x*y ~= hi*hi + hi*lo + lo*hi (f32 accumulate)
__device__ __forceinline__ v8f mm3(v16bf ahi, v16bf alo, v16bf bhi, v16bf blo, v8f acc) {
    acc = __builtin_amdgcn_wmma_f32_16x16x32_bf16(false, ahi, false, bhi,
                                                  (short)0, acc, false, false);
    acc = __builtin_amdgcn_wmma_f32_16x16x32_bf16(false, ahi, false, blo,
                                                  (short)0, acc, false, false);
    acc = __builtin_amdgcn_wmma_f32_16x16x32_bf16(false, alo, false, bhi,
                                                  (short)0, acc, false, false);
    return acc;
}

// ---------------------------------------------------------------------------
// Weight prep: src is row-major (Kdim x Ndim) fp32.  Emit bf16 hi/lo in
// n-major layout Bt[n*Kdim + k] so B-matrix WMMA fragments are contiguous
// 16-byte loads per lane.
// ---------------------------------------------------------------------------
__global__ __launch_bounds__(256)
void wprep_kernel(const float* __restrict__ src, __bf16* __restrict__ hi,
                  __bf16* __restrict__ lo, int Kdim, int Ndim) {
    int t = blockIdx.x * blockDim.x + threadIdx.x;
    if (t >= Kdim * Ndim) return;
    int n = t / Kdim;
    int k = t - n * Kdim;
    float v = src[(size_t)k * Ndim + n];
    __bf16 h, l; split_bf16(v, h, l);
    hi[t] = h; lo[t] = l;
}

// ---------------------------------------------------------------------------
// Depthwise 3x3 conv (SAME, zero pad) + bias; also split x itself to bf16.
// One thread per (n,h,w,c) element.
// ---------------------------------------------------------------------------
__global__ __launch_bounds__(256)
void dwconv_kernel(const float* __restrict__ x, const float* __restrict__ dww,
                   const float* __restrict__ dwb,
                   __bf16* __restrict__ xdw_hi, __bf16* __restrict__ xdw_lo,
                   __bf16* __restrict__ x_hi,  __bf16* __restrict__ x_lo) {
    int t = blockIdx.x * blockDim.x + threadIdx.x;
    if (t >= NHW * CC) return;
    int c   = t & (CC - 1);
    int nhw = t >> 8;
    int w   = nhw & (WW - 1);
    int h   = (nhw >> 6) & (HH - 1);
    int n   = nhw >> 12;

    float acc = dwb[c];
#pragma unroll
    for (int dy = -1; dy <= 1; ++dy) {
#pragma unroll
        for (int dx = -1; dx <= 1; ++dx) {
            int yy = h + dy, xx = w + dx;
            if (yy >= 0 && yy < HH && xx >= 0 && xx < WW) {
                acc += x[((size_t)((n * HH + yy) * WW + xx)) * CC + c] *
                       dww[((dy + 1) * 3 + (dx + 1)) * CC + c];
            }
        }
    }
    __bf16 hi, lo;
    split_bf16(acc, hi, lo);
    xdw_hi[t] = hi; xdw_lo[t] = lo;
    split_bf16(x[t], hi, lo);
    x_hi[t] = hi; x_lo[t] = lo;
}

// ---------------------------------------------------------------------------
// GEMM: C[M,Nout] = A[M,K] * B[K,Nout] + bias.  K multiple of 32,
// M multiple of 32, Nout multiple of 32.
// A row-major bf16 hi/lo; B n-major (Bt[n*K+k]) bf16 hi/lo.
// Each wave owns a 32x32 macro-tile: 4 accumulators, 8 fragment loads ->
// 12 v_wmma_f32_16x16x32_bf16 per 32-wide k-step (load:wmma = 0.67).
// ---------------------------------------------------------------------------
__global__ __launch_bounds__(128)
void gemm_bf16x3_kernel(const __bf16* __restrict__ Ah, const __bf16* __restrict__ Al,
                        const __bf16* __restrict__ Bh, const __bf16* __restrict__ Bl,
                        const float* __restrict__ bias, float* __restrict__ Cout,
                        int Mdim, int Kdim, int Ndim) {
    const int lane   = threadIdx.x & 31;
    const int wave   = threadIdx.x >> 5;
    const int tilesN2 = Ndim >> 5;                 // 32-wide n macro-tiles
    const int totalTiles = (Mdim >> 5) * tilesN2;  // 32-tall m macro-tiles
    int tile = blockIdx.x * (blockDim.x >> 5) + wave;
    if (tile >= totalTiles) return;
    const int tm = (tile / tilesN2) << 1;          // in 16-row units
    const int tn = (tile - (tile / tilesN2) * tilesN2) << 1;

    const int lr    = lane & 15;
    const int khalf = (lane >> 4) << 3;

    const __bf16* a0h = Ah + (size_t)(tm * 16 + lr) * Kdim + khalf;
    const __bf16* a0l = Al + (size_t)(tm * 16 + lr) * Kdim + khalf;
    const __bf16* a1h = a0h + (size_t)16 * Kdim;
    const __bf16* a1l = a0l + (size_t)16 * Kdim;
    const __bf16* b0h = Bh + (size_t)(tn * 16 + lr) * Kdim + khalf;
    const __bf16* b0l = Bl + (size_t)(tn * 16 + lr) * Kdim + khalf;
    const __bf16* b1h = b0h + (size_t)16 * Kdim;
    const __bf16* b1l = b0l + (size_t)16 * Kdim;

    v8f acc00 = {}, acc01 = {}, acc10 = {}, acc11 = {};
    for (int k0 = 0; k0 < Kdim; k0 += 32) {
        v16bf fa0h = load_frag(a0h + k0);
        v16bf fa0l = load_frag(a0l + k0);
        v16bf fa1h = load_frag(a1h + k0);
        v16bf fa1l = load_frag(a1l + k0);
        v16bf fb0h = load_frag(b0h + k0);
        v16bf fb0l = load_frag(b0l + k0);
        v16bf fb1h = load_frag(b1h + k0);
        v16bf fb1l = load_frag(b1l + k0);

        acc00 = mm3(fa0h, fa0l, fb0h, fb0l, acc00);
        acc01 = mm3(fa0h, fa0l, fb1h, fb1l, acc01);
        acc10 = mm3(fa1h, fa1l, fb0h, fb0l, acc10);
        acc11 = mm3(fa1h, fa1l, fb1h, fb1l, acc11);
    }

    // C/D layout: lanes 0-15 -> M = base + r, lanes 16-31 -> M = base + 8 + r,
    // N = base + (lane & 15)
    const int mhalf = (lane >> 4) << 3;
    const int n0 = tn * 16 + lr;
    const int n1 = n0 + 16;
    const int m0 = tm * 16 + mhalf;
    const int m1 = m0 + 16;
    const float bn0 = (bias != nullptr) ? bias[n0] : 0.0f;
    const float bn1 = (bias != nullptr) ? bias[n1] : 0.0f;
#pragma unroll
    for (int r = 0; r < 8; ++r) {
        Cout[(size_t)(m0 + r) * Ndim + n0] = acc00[r] + bn0;
        Cout[(size_t)(m0 + r) * Ndim + n1] = acc01[r] + bn1;
        Cout[(size_t)(m1 + r) * Ndim + n0] = acc10[r] + bn0;
        Cout[(size_t)(m1 + r) * Ndim + n1] = acc11[r] + bn1;
    }
}

// ---------------------------------------------------------------------------
// Deformable sampling: one thread per (n,h,w,g); handles GC=8 channels.
// Zero-padded bilinear gather of `value`, masked accumulate over 9 taps.
// Output written as bf16 hi/lo (input A of final GEMM).
// ---------------------------------------------------------------------------
__global__ __launch_bounds__(256)
void sample_kernel(const float* __restrict__ om, const float* __restrict__ value,
                   __bf16* __restrict__ s_hi, __bf16* __restrict__ s_lo) {
    int t = blockIdx.x * blockDim.x + threadIdx.x;
    if (t >= NHW * GG) return;
    int g   = t & (GG - 1);
    int nhw = t >> 5;
    int w   = nhw & (WW - 1);
    int h   = (nhw >> 6) & (HH - 1);
    int n   = nhw >> 12;

    const float* omr   = om + (size_t)nhw * OMN + g * 27;
    const float* vbase = value + ((size_t)n * HH * WW) * CC + g * GCH;

    float acc[GCH];
#pragma unroll
    for (int c = 0; c < GCH; ++c) acc[c] = 0.0f;

#pragma unroll
    for (int k = 0; k < 9; ++k) {
        const int ky = k / 3 - 1;
        const int kx = k % 3 - 1;
        const float offx = omr[2 * k];
        const float offy = omr[2 * k + 1];
        const float msk  = omr[18 + k];

        const float py = (float)(h + ky) + offy;
        const float px = (float)(w + kx) + offx;
        const float y0f = floorf(py);
        const float x0f = floorf(px);
        const float ty = py - y0f;
        const float tx = px - x0f;
        const int y0 = (int)y0f;
        const int x0 = (int)x0f;

        const float w00 = msk * (1.0f - ty) * (1.0f - tx);
        const float w01 = msk * (1.0f - ty) * tx;
        const float w10 = msk * ty * (1.0f - tx);
        const float w11 = msk * ty * tx;

        const int ys[2] = { y0, y0 + 1 };
        const int xs[2] = { x0, x0 + 1 };
        const float ws[2][2] = { { w00, w01 }, { w10, w11 } };

#pragma unroll
        for (int iy = 0; iy < 2; ++iy) {
#pragma unroll
            for (int ix = 0; ix < 2; ++ix) {
                const int yi = ys[iy], xi = xs[ix];
                const bool valid = (yi >= 0) && (yi < HH) && (xi >= 0) && (xi < WW);
                if (valid) {
                    const float* vp = vbase + ((size_t)yi * WW + xi) * CC;
                    const float wgt = ws[iy][ix];
#pragma unroll
                    for (int c = 0; c < GCH; ++c) acc[c] += wgt * vp[c];
                }
            }
        }
    }

    __bf16* ph = s_hi + (size_t)nhw * CC + g * GCH;
    __bf16* pl = s_lo + (size_t)nhw * CC + g * GCH;
#pragma unroll
    for (int c = 0; c < GCH; ++c) {
        __bf16 hi, lo;
        split_bf16(acc[c], hi, lo);
        ph[c] = hi; pl[c] = lo;
    }
}

// ---------------------------------------------------------------------------
// Host-side orchestration
// ---------------------------------------------------------------------------
static inline size_t align256(size_t x) { return (x + 255) & ~(size_t)255; }

extern "C" void kernel_launch(void* const* d_in, const int* in_sizes, int n_in,
                              void* d_out, int out_size, void* d_ws, size_t ws_size,
                              hipStream_t stream) {
    (void)in_sizes; (void)n_in; (void)out_size; (void)ws_size;

    const float* x    = (const float*)d_in[0];
    const float* dw_w = (const float*)d_in[1];
    const float* dw_b = (const float*)d_in[2];
    const float* om_w = (const float*)d_in[3];
    const float* om_b = (const float*)d_in[4];
    const float* vp_w = (const float*)d_in[5];
    const float* vp_b = (const float*)d_in[6];
    const float* op_w = (const float*)d_in[7];
    float* out = (float*)d_out;

    // Workspace carve-up
    char* ws = (char*)d_ws;
    size_t o = 0;
    const size_t actB = (size_t)NHW * CC * sizeof(__bf16);   // 16 MB per bf16 tensor

    __bf16* xdw_hi = (__bf16*)(ws + o); o = align256(o + actB);
    __bf16* xdw_lo = (__bf16*)(ws + o); o = align256(o + actB);
    __bf16* x_hi   = (__bf16*)(ws + o); o = align256(o + actB);
    __bf16* x_lo   = (__bf16*)(ws + o); o = align256(o + actB);
    float*  om     = (float*)(ws + o);  o = align256(o + (size_t)NHW * OMN * sizeof(float));
    float*  value  = (float*)(ws + o);  o = align256(o + (size_t)NHW * CC * sizeof(float));
    __bf16* s_hi   = (__bf16*)(ws + o); o = align256(o + actB);
    __bf16* s_lo   = (__bf16*)(ws + o); o = align256(o + actB);
    __bf16* omw_hi = (__bf16*)(ws + o); o = align256(o + (size_t)OMN * CC * sizeof(__bf16));
    __bf16* omw_lo = (__bf16*)(ws + o); o = align256(o + (size_t)OMN * CC * sizeof(__bf16));
    __bf16* vpw_hi = (__bf16*)(ws + o); o = align256(o + (size_t)CC * CC * sizeof(__bf16));
    __bf16* vpw_lo = (__bf16*)(ws + o); o = align256(o + (size_t)CC * CC * sizeof(__bf16));
    __bf16* opw_hi = (__bf16*)(ws + o); o = align256(o + (size_t)CC * CC * sizeof(__bf16));
    __bf16* opw_lo = (__bf16*)(ws + o); o = align256(o + (size_t)CC * CC * sizeof(__bf16));

    // 1. Weight transpose + bf16 split
    {
        int n1 = CC * OMN, n2 = CC * CC;
        wprep_kernel<<<(n1 + 255) / 256, 256, 0, stream>>>(om_w, omw_hi, omw_lo, CC, OMN);
        wprep_kernel<<<(n2 + 255) / 256, 256, 0, stream>>>(vp_w, vpw_hi, vpw_lo, CC, CC);
        wprep_kernel<<<(n2 + 255) / 256, 256, 0, stream>>>(op_w, opw_hi, opw_lo, CC, CC);
    }

    // 2. Depthwise conv + x split
    {
        int n = NHW * CC;
        dwconv_kernel<<<(n + 255) / 256, 256, 0, stream>>>(x, dw_w, dw_b,
                                                           xdw_hi, xdw_lo, x_hi, x_lo);
    }

    // 3. om = xdw @ om_w + om_b   (M=32768, K=256, N=864)
    {
        int tiles = (NHW / 32) * (OMN / 32);
        gemm_bf16x3_kernel<<<(tiles + 3) / 4, 128, 0, stream>>>(
            xdw_hi, xdw_lo, omw_hi, omw_lo, om_b, om, NHW, CC, OMN);
    }

    // 4. value = x @ vp_w + vp_b  (M=32768, K=256, N=256)
    {
        int tiles = (NHW / 32) * (CC / 32);
        gemm_bf16x3_kernel<<<(tiles + 3) / 4, 128, 0, stream>>>(
            x_hi, x_lo, vpw_hi, vpw_lo, vp_b, value, NHW, CC, CC);
    }

    // 5. Deformable bilinear sampling + mask
    {
        int n = NHW * GG;
        sample_kernel<<<(n + 255) / 256, 256, 0, stream>>>(om, value, s_hi, s_lo);
    }

    // 6. out = sampled @ op_w     (M=32768, K=256, N=256) -> d_out
    {
        int tiles = (NHW / 32) * (CC / 32);
        gemm_bf16x3_kernel<<<(tiles + 3) / 4, 128, 0, stream>>>(
            s_hi, s_lo, opw_hi, opw_lo, nullptr, out, NHW, CC, CC);
    }
}
